// MultiHeadAttention_20598663152115
// MI455X (gfx1250) — matrix operand order repaired and checked by reading, not compiled, not run on hardware
//
#include <hip/hip_runtime.h>
#include <hip/hip_bf16.h>
#include <math.h>

// ---------------------------------------------------------------------------
// MHA for MI455X (gfx1250): bf16 WMMA everywhere, fp32 accumulate.
//   D_MODEL=512, N_HEADS=8 (head_dim = 512), B=2, S=2048
// ---------------------------------------------------------------------------

typedef __attribute__((ext_vector_type(16))) __bf16 v16bf;
typedef __attribute__((ext_vector_type(8)))  __bf16 v8bf;
typedef __attribute__((ext_vector_type(8)))  float  v8f;
typedef __attribute__((ext_vector_type(4)))  float  v4f;

#define D_MODEL 512
#define N_HEADS 8
#define BATCH   2
#define SEQ     2048
#define DMX     (D_MODEL * N_HEADS)   // 4096
#define MROWS   (BATCH * SEQ)         // 4096

static __device__ __forceinline__ v8f wmma_bf16(v16bf a, v16bf b, v8f c) {
    // D = A(16x32 bf16) x B(32x16 bf16) + C(16x16 f32)
    return __builtin_amdgcn_wmma_f32_16x16x32_bf16(
        false, a, false, b, (short)0, c, false, false);
}

static __device__ __forceinline__ v16bf cat16(v8bf lo, v8bf hi) {
    return __builtin_shufflevector(lo, hi, 0, 1, 2, 3, 4, 5, 6, 7,
                                           8, 9, 10, 11, 12, 13, 14, 15);
}

// Pack two fp32 into adjacent bf16 and store as one b32.
static __device__ __forceinline__ void store_pk(__bf16* dst, float x, float y) {
    union { __bf16 h[2]; unsigned u; } p;
    p.h[0] = (__bf16)x; p.h[1] = (__bf16)y;
    *reinterpret_cast<unsigned*>(dst) = p.u;
}

// ---------------------------------------------------------------------------
// Kernel 1: fused QKV projection.  out = X @ W + bias, cast to bf16.
//   Tile: M=128, N=128, Kstep=32.  grid=(DMX/128, MROWS/128, 3), block=256.
//   Wave w owns N-cols [w*16, w*16+16) x all 8 M-subtiles -> 8 WMMA/K-step.
//   mode 0/1: store [B,H,S,D]      mode 2 (V): store [B,H,D,S]
// ---------------------------------------------------------------------------
__global__ __launch_bounds__(256)
void qkv_proj_kernel(const float* __restrict__ Q, const float* __restrict__ K,
                     const float* __restrict__ V,
                     const float* __restrict__ Wq, const float* __restrict__ bq,
                     const float* __restrict__ Wk, const float* __restrict__ bk,
                     const float* __restrict__ Wv, const float* __restrict__ bv,
                     __bf16* __restrict__ q_out, __bf16* __restrict__ k_out,
                     __bf16* __restrict__ vt_out) {
    __shared__ __bf16 At[128][32];    // 8 KB, [m][k] (k contiguous)
    __shared__ __bf16 BtT[128][32];   // 8 KB, [n][k] (k contiguous, transposed)

    const int mode = blockIdx.z;
    const float* X    = (mode == 0) ? Q  : (mode == 1) ? K  : V;
    const float* W    = (mode == 0) ? Wq : (mode == 1) ? Wk : Wv;
    const float* bias = (mode == 0) ? bq : (mode == 1) ? bk : bv;

    const int tid = threadIdx.x;
    const int m0 = blockIdx.y * 128;
    const int n0 = blockIdx.x * 128;
    const int wave = tid >> 5, lane = tid & 31;
    const int hi = lane >> 4, ln = lane & 15;

    v8f acc[8] = {};
    for (int kk = 0; kk < D_MODEL; kk += 32) {
        __syncthreads();
        // Stage A [128 x 32]: float2 loads along k, packed bf16 stores.
        #pragma unroll
        for (int j = 0; j < 8; ++j) {
            int i = tid + j * 256;            // pair index 0..2047
            int r = i >> 4, c2 = i & 15;
            float2 f = *reinterpret_cast<const float2*>(
                &X[(size_t)(m0 + r) * D_MODEL + kk + 2 * c2]);
            store_pk(&At[r][2 * c2], f.x, f.y);
        }
        // Stage B transposed [n][k]: coalesced global reads (consecutive n),
        // packed stores of two adjacent k values.
        #pragma unroll
        for (int j = 0; j < 8; ++j) {
            int i = tid + j * 256;            // pair index 0..2047
            int c = i & 127, r2 = i >> 7;     // n, k-pair
            float f0 = W[(size_t)(kk + 2 * r2)     * DMX + n0 + c];
            float f1 = W[(size_t)(kk + 2 * r2 + 1) * DMX + n0 + c];
            store_pk(&BtT[c][2 * r2], f0, f1);
        }
        __syncthreads();

        // B fragment: contiguous 32B from transposed tile.
        v16bf b = *reinterpret_cast<const v16bf*>(&BtT[wave * 16 + ln][0]);
        // Rearrange to contiguous k order: lane half hi holds K = hi*16 + e.
        v16bf bfrag;
        #pragma unroll
        for (int e = 0; e < 16; ++e) bfrag[e] = b[(hi * 16 + e) & 31];
        (void)bfrag;
        // NOTE: since K runs 0..31 contiguously in BtT, lane needs elements
        // [hi*16 .. hi*16+15]; load exactly that 32B window instead:
        v16bf bf2 = *reinterpret_cast<const v16bf*>(&BtT[wave * 16 + ln][0]);
        (void)bf2;
        v16bf bb;
        {
            const __bf16* bp = &BtT[wave * 16 + ln][hi * 16];
            v8bf l0 = *reinterpret_cast<const v8bf*>(bp);
            v8bf l1 = *reinterpret_cast<const v8bf*>(bp + 8);
            bb = cat16(l0, l1);
        }
        #pragma unroll
        for (int mt = 0; mt < 8; ++mt) {
            const __bf16* ap = &At[mt * 16 + ln][0];
            v8bf lo = *reinterpret_cast<const v8bf*>(ap + hi * 16);
            v8bf hh = *reinterpret_cast<const v8bf*>(ap + hi * 16 + 8);
            acc[mt] = wmma_bf16(cat16(lo, hh), bb, acc[mt]);
        }
    }

    // Epilogue: bias + scatter to head-split layouts.
    const int nb = n0 + wave * 16 + ln;
    const float bv_ = bias[nb];
    const int h = nb >> 9, d = nb & 511;
    #pragma unroll
    for (int mt = 0; mt < 8; ++mt) {
        #pragma unroll
        for (int r = 0; r < 8; ++r) {
            int m = m0 + mt * 16 + r + 8 * hi;
            int bi = m >> 11, s = m & 2047;
            float val = acc[mt][r] + bv_;
            if (mode == 2) {
                vt_out[(((size_t)bi * N_HEADS + h) * D_MODEL + d) * SEQ + s] = (__bf16)val;
            } else {
                __bf16* dst = (mode == 0) ? q_out : k_out;
                dst[(((size_t)bi * N_HEADS + h) * SEQ + s) * D_MODEL + d] = (__bf16)val;
            }
        }
    }
}

// ---------------------------------------------------------------------------
// Kernel 2: attention, one block per (b, h, 16-query-row tile).
//   Q fragments hoisted into 128 VGPRs; full 16x2048 fp32 score block in LDS
//   (128 KB of the WGP's 320 KB) -> exact one-pass softmax.
//   grid = (SEQ/16, N_HEADS, BATCH), block = 256 (8 waves)
// ---------------------------------------------------------------------------
__global__ __launch_bounds__(256)
void attn_kernel(const __bf16* __restrict__ q, const __bf16* __restrict__ k,
                 const __bf16* __restrict__ vt, __bf16* __restrict__ ao) {
    __shared__ float Sc[16][SEQ];        // 128 KB score block (later holds P)

    const int tid = threadIdx.x;
    const int s0 = blockIdx.x * 16;
    const int h  = blockIdx.y, bi = blockIdx.z;
    const int wave = tid >> 5, lane = tid & 31;
    const int hi = lane >> 4, ln = lane & 15;

    const __bf16* qh = q  + (size_t)(bi * N_HEADS + h) * SEQ * D_MODEL;
    const __bf16* kh = k  + (size_t)(bi * N_HEADS + h) * SEQ * D_MODEL;
    const __bf16* vh = vt + (size_t)(bi * N_HEADS + h) * D_MODEL * SEQ;

    // ---- Phase 0: hoist all 16 Q fragments (full K=512) into registers ----
    v16bf qf[16];
    {
        const __bf16* qp = &qh[(size_t)(s0 + ln) * D_MODEL];
        #pragma unroll
        for (int t = 0; t < 16; ++t) {
            v8bf lo = *reinterpret_cast<const v8bf*>(qp + t * 32 + hi * 16);
            v8bf hh = *reinterpret_cast<const v8bf*>(qp + t * 32 + hi * 16 + 8);
            qf[t] = cat16(lo, hh);
        }
    }

    const float scale = 0.044194173824159216f;   // 1/sqrt(512)

    // ---- Phase 1: scores = Q . K^T * scale  (pure load+WMMA loop) --------
    for (int chunk = wave; chunk < SEQ / 16; chunk += 8) {
        const __bf16* kp = kh + (size_t)(chunk * 16 + ln) * D_MODEL + hi * 16;
        if (chunk + 8 < SEQ / 16)                 // -> global_prefetch_b8
            __builtin_prefetch(kh + (size_t)((chunk + 8) * 16 + ln) * D_MODEL, 0, 1);
        v8f acc = {};
        #pragma unroll
        for (int t = 0; t < 16; ++t) {
            v16bf b = *reinterpret_cast<const v16bf*>(kp + t * 32);  // 32B
            acc = wmma_bf16(qf[t], b, acc);
        }
        #pragma unroll
        for (int r = 0; r < 8; ++r)
            Sc[r + 8 * hi][chunk * 16 + ln] = acc[r] * scale;
    }
    __syncthreads();

    // ---- Phase 2: row softmax (16 lanes per row, shfl_xor reductions) ----
    {
        const int r = tid >> 4, c0 = tid & 15;
        float mx = -3.4e38f;
        for (int j = 0; j < SEQ / 16; ++j)
            mx = fmaxf(mx, Sc[r][c0 + 16 * j]);
        #pragma unroll
        for (int off = 1; off < 16; off <<= 1)
            mx = fmaxf(mx, __shfl_xor(mx, off, 16));
        float sum = 0.f;
        for (int j = 0; j < SEQ / 16; ++j) {
            int c = c0 + 16 * j;
            float e = __expf(Sc[r][c] - mx);
            Sc[r][c] = e;
            sum += e;
        }
        #pragma unroll
        for (int off = 1; off < 16; off <<= 1)
            sum += __shfl_xor(sum, off, 16);
        float inv = 1.0f / sum;
        for (int j = 0; j < SEQ / 16; ++j)
            Sc[r][c0 + 16 * j] *= inv;
    }
    __syncthreads();

    // ---- Phase 3: out = P . V  (wave owns 64 head-dim cols) --------------
    const int nb0 = wave * 64;
    v8f oacc[4] = {};
    for (int kk = 0; kk < SEQ; kk += 32) {
        // A fragment: 4x ds_load_b128 of fp32 P, packed-convert to bf16.
        v4f p0 = *reinterpret_cast<const v4f*>(&Sc[ln][kk + hi * 16]);
        v4f p1 = *reinterpret_cast<const v4f*>(&Sc[ln][kk + hi * 16 + 4]);
        v4f p2 = *reinterpret_cast<const v4f*>(&Sc[ln][kk + hi * 16 + 8]);
        v4f p3 = *reinterpret_cast<const v4f*>(&Sc[ln][kk + hi * 16 + 12]);
        v16bf a;
        #pragma unroll
        for (int e = 0; e < 4; ++e) {
            a[e]      = (__bf16)p0[e];
            a[4 + e]  = (__bf16)p1[e];
            a[8 + e]  = (__bf16)p2[e];
            a[12 + e] = (__bf16)p3[e];
        }
        #pragma unroll
        for (int nt = 0; nt < 4; ++nt) {
            const __bf16* vp = vh + (size_t)(nb0 + nt * 16 + ln) * SEQ + kk + hi * 16;
            v16bf b = *reinterpret_cast<const v16bf*>(vp);   // 32B contiguous
            oacc[nt] = wmma_bf16(a, b, oacc[nt]);
        }
    }
    // Store out tile to [B, S, H*D] (bf16) for the output projection.
    #pragma unroll
    for (int nt = 0; nt < 4; ++nt) {
        #pragma unroll
        for (int r = 0; r < 8; ++r) {
            int s = s0 + r + 8 * hi;
            int col = h * D_MODEL + nb0 + nt * 16 + ln;
            ao[((size_t)(bi * SEQ + s)) * DMX + col] = (__bf16)oacc[nt][r];
        }
    }
}

// ---------------------------------------------------------------------------
// Kernel 3: output projection.  out = AO @ Wo + bo (fp32 out).
//   Tile: M=128, N=128, Kstep=32.  grid=(D_MODEL/128, MROWS/128), block=256.
// ---------------------------------------------------------------------------
__global__ __launch_bounds__(256)
void out_proj_kernel(const __bf16* __restrict__ AO, const float* __restrict__ Wo,
                     const float* __restrict__ bo, float* __restrict__ out) {
    __shared__ __bf16 At[128][32];    // 8 KB, [m][k]
    __shared__ __bf16 BtT[128][32];   // 8 KB, [n][k] transposed

    const int tid = threadIdx.x;
    const int m0 = blockIdx.y * 128;
    const int n0 = blockIdx.x * 128;
    const int wave = tid >> 5, lane = tid & 31;
    const int hi = lane >> 4, ln = lane & 15;

    v8f acc[8] = {};
    for (int kk = 0; kk < DMX; kk += 32) {
        __syncthreads();
        // Stage A [128 x 32]: bf16 pairs copied as b32.
        #pragma unroll
        for (int j = 0; j < 8; ++j) {
            int i = tid + j * 256;
            int r = i >> 4, c2 = i & 15;
            *reinterpret_cast<unsigned*>(&At[r][2 * c2]) =
                *reinterpret_cast<const unsigned*>(
                    &AO[(size_t)(m0 + r) * DMX + kk + 2 * c2]);
        }
        // Stage Wo transposed [n][k], fp32 -> packed bf16.
        #pragma unroll
        for (int j = 0; j < 8; ++j) {
            int i = tid + j * 256;
            int c = i & 127, r2 = i >> 7;
            float f0 = Wo[(size_t)(kk + 2 * r2)     * D_MODEL + n0 + c];
            float f1 = Wo[(size_t)(kk + 2 * r2 + 1) * D_MODEL + n0 + c];
            store_pk(&BtT[c][2 * r2], f0, f1);
        }
        __syncthreads();

        v16bf bb;
        {
            const __bf16* bp = &BtT[wave * 16 + ln][hi * 16];
            v8bf l0 = *reinterpret_cast<const v8bf*>(bp);
            v8bf l1 = *reinterpret_cast<const v8bf*>(bp + 8);
            bb = cat16(l0, l1);
        }
        #pragma unroll
        for (int mt = 0; mt < 8; ++mt) {
            const __bf16* ap = &At[mt * 16 + ln][0];
            v8bf lo = *reinterpret_cast<const v8bf*>(ap + hi * 16);
            v8bf hh = *reinterpret_cast<const v8bf*>(ap + hi * 16 + 8);
            acc[mt] = wmma_bf16(cat16(lo, hh), bb, acc[mt]);
        }
    }

    const int nb = n0 + wave * 16 + ln;
    const float bv_ = bo[nb];
    #pragma unroll
    for (int mt = 0; mt < 8; ++mt) {
        #pragma unroll
        for (int r = 0; r < 8; ++r)
            out[(size_t)(m0 + mt * 16 + r + 8 * hi) * D_MODEL + nb] =
                acc[mt][r] + bv_;
    }
}

// ---------------------------------------------------------------------------
// Host-side launch.
// ---------------------------------------------------------------------------
extern "C" void kernel_launch(void* const* d_in, const int* in_sizes, int n_in,
                              void* d_out, int out_size, void* d_ws, size_t ws_size,
                              hipStream_t stream) {
    const float* Q  = (const float*)d_in[0];
    const float* K  = (const float*)d_in[1];
    const float* V  = (const float*)d_in[2];
    const float* Wq = (const float*)d_in[3];
    const float* bq = (const float*)d_in[4];
    const float* Wk = (const float*)d_in[5];
    const float* bk = (const float*)d_in[6];
    const float* Wv = (const float*)d_in[7];
    const float* bv = (const float*)d_in[8];
    const float* Wo = (const float*)d_in[9];
    const float* bo = (const float*)d_in[10];
    float* out = (float*)d_out;

    // Workspace (bf16): q [B,H,S,D] | k [B,H,S,D] | vT [B,H,D,S] | ao [B,S,H*D]
    const size_t per = (size_t)BATCH * N_HEADS * SEQ * D_MODEL;   // 16.7M elems
    __bf16* q_ws  = (__bf16*)d_ws;
    __bf16* k_ws  = q_ws  + per;
    __bf16* vt_ws = k_ws  + per;
    __bf16* ao_ws = vt_ws + per;

    qkv_proj_kernel<<<dim3(DMX / 128, MROWS / 128, 3), 256, 0, stream>>>(
        Q, K, V, Wq, bq, Wk, bk, Wv, bv, q_ws, k_ws, vt_ws);

    attn_kernel<<<dim3(SEQ / 16, N_HEADS, BATCH), 256, 0, stream>>>(
        q_ws, k_ws, vt_ws, ao_ws);

    out_proj_kernel<<<dim3(D_MODEL / 128, MROWS / 128), 256, 0, stream>>>(
        ao_ws, Wo, bo, out);
}
